// GroupedQueryAttention_36369783063125
// MI455X (gfx1250) — matrix operand order repaired and checked by reading, not compiled, not run on hardware
//
#include <hip/hip_runtime.h>
#include <hip/hip_bf16.h>
#include <math.h>

// ---------------------------------------------------------------------------
// GQA attention block for MI455X (gfx1250, wave32, WMMA).
// Pipeline: cast->bf16, WMMA QKV GEMMs, RoPE+layout, flash attention (WMMA),
// WMMA output projection. All matmuls: v_wmma_f32_16x16x32_bf16, fp32 accum.
// ---------------------------------------------------------------------------

typedef __attribute__((ext_vector_type(16))) __bf16 bf16x16;
typedef __attribute__((ext_vector_type(8)))  __bf16 bf16x8;
typedef __attribute__((ext_vector_type(8)))  float  f32x8;

#define B_   2
#define S_   2048
#define D_   2048
#define H_   16
#define G_   4
#define HD_  128
#define KV_  (G_*HD_)      // 512
#define M_   (B_*S_)       // 4096

__device__ __forceinline__ bf16x8 ld8(const __bf16* p) {
    return *(const bf16x8*)p;
}
__device__ __forceinline__ bf16x16 frag2(bf16x8 a, bf16x8 b) {
    bf16x16 r;
#pragma unroll
    for (int i = 0; i < 8; ++i) { r[i] = a[i]; r[i + 8] = b[i]; }
    return r;
}
__device__ __forceinline__ f32x8 wmma_bf16(bf16x16 a, bf16x16 b, f32x8 c) {
    return __builtin_amdgcn_wmma_f32_16x16x32_bf16(
        /*neg_a=*/false, a, /*neg_b=*/false, b,
        /*c_mod=*/(short)0, c, /*reuse_a=*/false, /*reuse_b=*/false);
}

// ---------------------------------------------------------------------------
// fp32 -> bf16 cast
// ---------------------------------------------------------------------------
__global__ void cast_bf16_kernel(const float* __restrict__ in,
                                 __bf16* __restrict__ out, long n) {
    long i = (long)blockIdx.x * blockDim.x + threadIdx.x;
    if (i < n) out[i] = (__bf16)in[i];
}

// ---------------------------------------------------------------------------
// C[M x N] (fp32) = A[M x K](bf16, row major) @ W[N x K](bf16, row major)^T
// Block: 128 threads = 4 waves; block tile 64x128; wave tile 32x64; K-step 32.
// A fragment: lane row = m + (lane&15); K = k0+16*hl .. +15 (contig).
// B fragment: lane col = n + (lane&15) -> W row; K = k0+16*hl .. +15 (contig).
// ---------------------------------------------------------------------------
__global__ __launch_bounds__(128)
void wgemm_kernel(const __bf16* __restrict__ A, const __bf16* __restrict__ W,
                  float* __restrict__ C, int M, int N, int K) {
    const int lane = threadIdx.x & 31;
    const int wave = threadIdx.x >> 5;
    const int hl = lane >> 4;
    const int lr = lane & 15;
    const int m0 = blockIdx.x * 64 + (wave & 1) * 32;
    const int n0 = blockIdx.y * 128 + (wave >> 1) * 64;

    f32x8 acc[2][4];
#pragma unroll
    for (int i = 0; i < 2; ++i)
#pragma unroll
        for (int j = 0; j < 4; ++j) acc[i][j] = (f32x8)0.0f;

    const __bf16* ar0 = A + (long)(m0 + lr) * K;
    const __bf16* ar1 = A + (long)(m0 + 16 + lr) * K;

    for (int k0 = 0; k0 < K; k0 += 32) {
        bf16x16 a0 = frag2(ld8(ar0 + k0 + 16 * hl), ld8(ar0 + k0 + 16 * hl + 8));
        bf16x16 a1 = frag2(ld8(ar1 + k0 + 16 * hl), ld8(ar1 + k0 + 16 * hl + 8));
#pragma unroll
        for (int j = 0; j < 4; ++j) {
            const __bf16* wr = W + (long)(n0 + j * 16 + lr) * K + k0 + 16 * hl;
            bf16x16 b = frag2(ld8(wr), ld8(wr + 8));
            acc[0][j] = wmma_bf16(a0, b, acc[0][j]);
            acc[1][j] = wmma_bf16(a1, b, acc[1][j]);
        }
    }
#pragma unroll
    for (int mi = 0; mi < 2; ++mi)
#pragma unroll
        for (int j = 0; j < 4; ++j)
#pragma unroll
            for (int r = 0; r < 8; ++r) {
                int m = m0 + mi * 16 + r + 8 * hl;
                int n = n0 + j * 16 + lr;
                C[(long)m * N + n] = acc[mi][j][r];
            }
}

// ---------------------------------------------------------------------------
// RoPE (first 64 dims, interleaved pairs) + cast + layout change:
// src fp32 [B*S][NH*128] (col = h*128+d) -> dst bf16 [(b*NH+h)][s][128]
// one thread per (row, h, pair j in 0..63)
// ---------------------------------------------------------------------------
__global__ void rope_cast_kernel(const float* __restrict__ src,
                                 __bf16* __restrict__ dst, int NH, long total) {
    long idx = (long)blockIdx.x * blockDim.x + threadIdx.x;
    if (idx >= total) return;
    int j = (int)(idx & 63);
    long t = idx >> 6;
    int h = (int)(t % NH);
    long row = t / NH;                  // b*S + s
    int s = (int)(row % S_);
    long b = row / S_;
    const float* p = src + row * ((long)NH * HD_) + (long)h * HD_;
    __bf16* o = dst + ((b * NH + h) * (long)S_ + s) * HD_;
    float a = p[2 * j], bb = p[2 * j + 1];
    if (j < 32) {
        // inv = theta^(-2j/64) = exp(-(2j/64)*ln(10000))
        float inv = __expf(-((float)(2 * j) / 64.0f) * 9.210340371976184f);
        float ang = (float)s * inv;
        float sn, cs;
        __sincosf(ang, &sn, &cs);
        o[2 * j]     = (__bf16)(a * cs - bb * sn);
        o[2 * j + 1] = (__bf16)(a * sn + bb * cs);
    } else {
        o[2 * j]     = (__bf16)a;
        o[2 * j + 1] = (__bf16)bb;
    }
}

// ---------------------------------------------------------------------------
// V transpose + cast: Vf32 [B*S][512] (col = g*128+d) -> Vt bf16 [(b*4+g)][128][S]
// ---------------------------------------------------------------------------
__global__ void vtrans_kernel(const float* __restrict__ V,
                              __bf16* __restrict__ Vt, long total) {
    long idx = (long)blockIdx.x * blockDim.x + threadIdx.x;
    if (idx >= total) return;
    int c = (int)(idx % KV_);
    long row = idx / KV_;               // b*S + s
    int s = (int)(row % S_);
    long b = row / S_;
    int g = c >> 7, d = c & 127;
    Vt[((b * G_ + g) * (long)HD_ + d) * S_ + s] = (__bf16)V[idx];
}

// ---------------------------------------------------------------------------
// Flash attention: block = 128 thr = 4 waves; wave owns 16 query rows.
// Q: bf16 [(b*16+h)][s][128]   K: bf16 [(b*4+g)][s][128]   Vt: bf16 [(b*4+g)][128][S]
// O: bf16 [b*S+s][2048] (col = h*128+d)  -- ready for the output projection.
// Per 32-key iteration: 8 score WMMAs + online softmax + LDS P-redistribute
// + 8 PV WMMAs.
// ---------------------------------------------------------------------------
__global__ __launch_bounds__(128)
void attn_kernel(const __bf16* __restrict__ Q, const __bf16* __restrict__ Kc,
                 const __bf16* __restrict__ Vt, __bf16* __restrict__ O) {
    __shared__ __bf16 Pl[4][16 * 32];
    const int lane = threadIdx.x & 31;
    const int wave = threadIdx.x >> 5;
    const int hl = lane >> 4;
    const int lr = lane & 15;
    const int bh = blockIdx.y;          // b*16 + h
    const int b = bh >> 4, h = bh & 15;
    const int g = h >> 2;               // kv group (rep = 4)
    const int q0 = blockIdx.x * 64 + wave * 16;

    const __bf16* qrow  = Q  + ((long)bh * S_ + q0 + lr) * HD_;
    const __bf16* kbase = Kc + ((long)(b * G_ + g) * S_) * HD_;
    const __bf16* vbase = Vt + ((long)(b * G_ + g) * HD_) * (long)S_;

    const float scale = 0.08838834764831845f;  // 1/sqrt(128)

    float m8[8], l8[8];
    f32x8 o[8];
#pragma unroll
    for (int r = 0; r < 8; ++r) { m8[r] = -INFINITY; l8[r] = 0.0f; }
#pragma unroll
    for (int j = 0; j < 8; ++j) o[j] = (f32x8)0.0f;

    // Q fragments for all 4 d-chunks (K-dim of the score GEMM)
    bf16x16 qf[4];
#pragma unroll
    for (int d0 = 0; d0 < 4; ++d0)
        qf[d0] = frag2(ld8(qrow + d0 * 32 + 16 * hl),
                       ld8(qrow + d0 * 32 + 16 * hl + 8));

    __bf16* pl = Pl[wave];

    for (int kb = 0; kb <= q0 + 15; kb += 32) {
        // ---- scores: 16 queries x 32 keys, K-dim = 128 head dims
        f32x8 c0 = (f32x8)0.0f, c1 = (f32x8)0.0f;
#pragma unroll
        for (int d0 = 0; d0 < 4; ++d0) {
            const __bf16* kr0 = kbase + (long)(kb + lr) * HD_ + d0 * 32 + 16 * hl;
            const __bf16* kr1 = kbase + (long)(kb + 16 + lr) * HD_ + d0 * 32 + 16 * hl;
            bf16x16 b0 = frag2(ld8(kr0), ld8(kr0 + 8));
            bf16x16 b1 = frag2(ld8(kr1), ld8(kr1 + 8));
            c0 = wmma_bf16(qf[d0], b0, c0);
            c1 = wmma_bf16(qf[d0], b1, c1);
        }
        // ---- scale + causal mask (element: row = q0+r+8*hl, key = kb(+16)+lr)
        float s0[8], s1[8];
#pragma unroll
        for (int r = 0; r < 8; ++r) {
            int qr = q0 + r + 8 * hl;
            s0[r] = (kb + lr > qr)      ? -INFINITY : c0[r] * scale;
            s1[r] = (kb + 16 + lr > qr) ? -INFINITY : c1[r] * scale;
        }
        // ---- online softmax (row-wise reductions across 16-lane halves)
        float p0[8], p1[8];
#pragma unroll
        for (int r = 0; r < 8; ++r) {
            float rm = fmaxf(s0[r], s1[r]);
            rm = fmaxf(rm, __shfl_xor(rm, 1, 32));
            rm = fmaxf(rm, __shfl_xor(rm, 2, 32));
            rm = fmaxf(rm, __shfl_xor(rm, 4, 32));
            rm = fmaxf(rm, __shfl_xor(rm, 8, 32));
            float mn = fmaxf(m8[r], rm);
            float alpha = __expf(m8[r] - mn);
            float e0 = __expf(s0[r] - mn);
            float e1 = __expf(s1[r] - mn);
            float rs = e0 + e1;
            rs += __shfl_xor(rs, 1, 32);
            rs += __shfl_xor(rs, 2, 32);
            rs += __shfl_xor(rs, 4, 32);
            rs += __shfl_xor(rs, 8, 32);
            l8[r] = l8[r] * alpha + rs;
            m8[r] = mn;
#pragma unroll
            for (int j = 0; j < 8; ++j) o[j][r] *= alpha;
            p0[r] = e0;
            p1[r] = e1;
        }
        // ---- P (C-frag layout) -> LDS -> A-frag layout
#pragma unroll
        for (int r = 0; r < 8; ++r) {
            int row = r + 8 * hl;
            pl[row * 32 + lr]      = (__bf16)p0[r];
            pl[row * 32 + 16 + lr] = (__bf16)p1[r];
        }
        asm volatile("" ::: "memory");  // keep ds stores before ds loads
        bf16x16 pa = frag2(ld8(pl + lr * 32 + 16 * hl),
                           ld8(pl + lr * 32 + 16 * hl + 8));
        // ---- P @ V : 8 head-dim tiles of 16
#pragma unroll
        for (int j = 0; j < 8; ++j) {
            const __bf16* vr = vbase + (long)(j * 16 + lr) * S_ + kb + 16 * hl;
            bf16x16 bv = frag2(ld8(vr), ld8(vr + 8));
            o[j] = wmma_bf16(pa, bv, o[j]);
        }
    }
    // ---- normalize and store (col = h*128 + j*16 + lr)
#pragma unroll
    for (int j = 0; j < 8; ++j)
#pragma unroll
        for (int r = 0; r < 8; ++r) {
            long row = (long)b * S_ + q0 + r + 8 * hl;
            float v = o[j][r] / l8[r];
            O[row * D_ + h * HD_ + j * 16 + lr] = (__bf16)v;
        }
}

// ---------------------------------------------------------------------------
// Host side
// ---------------------------------------------------------------------------
static inline long cdiv(long a, long b) { return (a + b - 1) / b; }

extern "C" void kernel_launch(void* const* d_in, const int* in_sizes, int n_in,
                              void* d_out, int out_size, void* d_ws, size_t ws_size,
                              hipStream_t stream) {
    (void)in_sizes; (void)n_in; (void)out_size; (void)ws_size;
    const float* x  = (const float*)d_in[0];
    const float* wq = (const float*)d_in[1];
    const float* wk = (const float*)d_in[2];
    const float* wv = (const float*)d_in[3];
    const float* wo = (const float*)d_in[4];
    float* out = (float*)d_out;

    char* ws = (char*)d_ws;
    size_t off = 0;
    auto alloc = [&](size_t bytes) -> char* {
        char* p = ws + off;
        off = (off + bytes + 255) & ~(size_t)255;
        return p;
    };
    __bf16* X16  = (__bf16*)alloc((size_t)M_ * D_ * 2);          // 16 MB
    __bf16* Wq16 = (__bf16*)alloc((size_t)D_ * D_ * 2);          //  8 MB
    __bf16* Wk16 = (__bf16*)alloc((size_t)KV_ * D_ * 2);         //  2 MB
    __bf16* Wv16 = (__bf16*)alloc((size_t)KV_ * D_ * 2);         //  2 MB
    __bf16* Wo16 = (__bf16*)alloc((size_t)D_ * D_ * 2);          //  8 MB
    float*  Qf   = (float*)alloc((size_t)M_ * D_ * 4);           // 32 MB
    float*  Kf   = (float*)alloc((size_t)M_ * KV_ * 4);          //  8 MB
    float*  Vf   = (float*)alloc((size_t)M_ * KV_ * 4);          //  8 MB
    __bf16* Q16  = (__bf16*)alloc((size_t)B_ * H_ * S_ * HD_ * 2); // 16 MB
    __bf16* K16  = (__bf16*)alloc((size_t)B_ * G_ * S_ * HD_ * 2); //  4 MB
    __bf16* Vt16 = (__bf16*)alloc((size_t)B_ * G_ * HD_ * S_ * 2); //  4 MB
    __bf16* O16  = (__bf16*)alloc((size_t)M_ * D_ * 2);          // 16 MB

    const int CT = 256;
    // 1) casts
    { long n = (long)M_ * D_;   cast_bf16_kernel<<<cdiv(n, CT), CT, 0, stream>>>(x,  X16,  n); }
    { long n = (long)D_ * D_;   cast_bf16_kernel<<<cdiv(n, CT), CT, 0, stream>>>(wq, Wq16, n); }
    { long n = (long)KV_ * D_;  cast_bf16_kernel<<<cdiv(n, CT), CT, 0, stream>>>(wk, Wk16, n); }
    { long n = (long)KV_ * D_;  cast_bf16_kernel<<<cdiv(n, CT), CT, 0, stream>>>(wv, Wv16, n); }
    { long n = (long)D_ * D_;   cast_bf16_kernel<<<cdiv(n, CT), CT, 0, stream>>>(wo, Wo16, n); }

    // 2) QKV projections (WMMA)
    wgemm_kernel<<<dim3(M_ / 64, D_ / 128),  128, 0, stream>>>(X16, Wq16, Qf, M_, D_,  D_);
    wgemm_kernel<<<dim3(M_ / 64, KV_ / 128), 128, 0, stream>>>(X16, Wk16, Kf, M_, KV_, D_);
    wgemm_kernel<<<dim3(M_ / 64, KV_ / 128), 128, 0, stream>>>(X16, Wv16, Vf, M_, KV_, D_);

    // 3) RoPE + layout + cast
    { long n = (long)M_ * H_ * 64; rope_cast_kernel<<<cdiv(n, CT), CT, 0, stream>>>(Qf, Q16, H_, n); }
    { long n = (long)M_ * G_ * 64; rope_cast_kernel<<<cdiv(n, CT), CT, 0, stream>>>(Kf, K16, G_, n); }
    { long n = (long)M_ * KV_;     vtrans_kernel<<<cdiv(n, CT), CT, 0, stream>>>(Vf, Vt16, n); }

    // 4) flash attention (WMMA)
    attn_kernel<<<dim3(S_ / 64, B_ * H_), 128, 0, stream>>>(Q16, K16, Vt16, O16);

    // 5) output projection (WMMA) -> fp32 d_out
    wgemm_kernel<<<dim3(M_ / 64, D_ / 128), 128, 0, stream>>>(O16, Wo16, out, M_, D_, D_);
}
